// bnLSTM_17583596110117
// MI455X (gfx1250) — matrix operand order repaired and checked by reading, not compiled, not run on hardware
//
#include <hip/hip_runtime.h>
#include <hip/hip_bf16.h>
#include <cstdint>

// ---------------------------------------------------------------------------
// bnLSTM for MI455X (gfx1250): conv1d(im2col WMMA GEMM + fused pool/relu)
// -> input-projection WMMA GEMM -> 256 sequential LSTM step kernels with
// wave-local batch-norm -> FC+softmax.  Matrix math on
// v_wmma_f32_16x16x32_bf16 (wave32); LDS staging via async global->LDS
// (ASYNCcnt) with double-buffered K-chunks.
// ---------------------------------------------------------------------------

typedef __attribute__((ext_vector_type(16))) __bf16 v16bf;
typedef __attribute__((ext_vector_type(8)))  float  v8f;

union VB16 { v16bf v; uint4 u[2]; };

#define WMMA_BF16(A, B, C) \
  __builtin_amdgcn_wmma_f32_16x16x32_bf16(false, (A), false, (B), (short)0, (C), false, false)

// Async global -> LDS copy of 16 bytes (per-lane). Tracked by ASYNCcnt.
__device__ __forceinline__ void async_load_b128(void* lds_dst, const void* gaddr) {
  uint32_t lds_off = (uint32_t)(uintptr_t)lds_dst;   // low 32 bits = LDS byte offset
  asm volatile("global_load_async_to_lds_b128 %0, %1, off"
               :: "v"(lds_off), "v"(gaddr) : "memory");
}
__device__ __forceinline__ void wait_async0() {
  asm volatile("s_wait_asynccnt 0x0" ::: "memory");
}

// A fragment (16x32 bf16, row-major in LDS, strideHalf halfs per row).
// CDNA5 layout: lanes 0-15 row=ln, K {0..7,16..23}; lanes 16-31 row=ln, K {8..15,24..31}.
__device__ __forceinline__ v16bf lds_loadA(const __bf16* tileBase, int strideHalf, int ln, int hl) {
  VB16 r;
  const __bf16* p0 = tileBase + ln * strideHalf + hl * 8;
  r.u[0] = *reinterpret_cast<const uint4*>(p0);
  r.u[1] = *reinterpret_cast<const uint4*>(p0 + 16);
  return r.v;
}

// B fragment (32x16 bf16) from a K-contiguous (pre-transposed) weight:
// Bt[col*Kdim + k]. Lanes 0-15: col, K {0..7,16..23}; lanes 16-31: col, K {8..15,24..31}.
__device__ __forceinline__ v16bf glb_loadB(const __bf16* Bt, int Kdim, int col, int k0, int hl) {
  VB16 r;
  const __bf16* p = Bt + (size_t)col * Kdim + k0 + hl * 8;
  r.u[0] = *reinterpret_cast<const uint4*>(p);
  r.u[1] = *reinterpret_cast<const uint4*>(p + 16);
  return r.v;
}

__device__ __forceinline__ v8f v8f_zero() {
  v8f z;
#pragma unroll
  for (int e = 0; e < 8; ++e) z[e] = 0.0f;
  return z;
}

__device__ __forceinline__ float sigmoidf_(float x) { return 1.0f / (1.0f + __expf(-x)); }

// ---------------------------------------------------------------------------
// Prep kernels
// ---------------------------------------------------------------------------
__global__ __launch_bounds__(256) void k_cvt_x(const float* __restrict__ x, __bf16* __restrict__ xbf, int n) {
  int i = blockIdx.x * 256 + threadIdx.x;
  if (i < n) xbf[i] = (__bf16)x[i];
}

// conv_w (512,64,33) -> Bconv[co*2112 + k*64+ci]  (K=k*64+ci contiguous per col)
__global__ __launch_bounds__(256) void k_pack_convw(const float* __restrict__ w, __bf16* __restrict__ Bt) {
  int i = blockIdx.x * 256 + threadIdx.x;
  if (i >= 512 * 2112) return;
  int co = i / 2112, kk = i % 2112;
  int k = kk >> 6, ci = kk & 63;
  Bt[i] = (__bf16)w[co * 2112 + ci * 33 + k];
}

// W (512, 2048) row-major -> Wt[col*512 + k]
__global__ __launch_bounds__(256) void k_pack_wt(const float* __restrict__ w, __bf16* __restrict__ Bt) {
  int i = blockIdx.x * 256 + threadIdx.x;
  if (i >= 2048 * 512) return;
  int col = i / 512, k = i % 512;
  Bt[i] = (__bf16)w[k * 2048 + col];
}

__global__ __launch_bounds__(256) void k_init_hc(const float* __restrict__ h0, const float* __restrict__ c0,
                                                 __bf16* __restrict__ hb, float* __restrict__ hf,
                                                 float* __restrict__ c) {
  int i = blockIdx.x * 256 + threadIdx.x;
  if (i >= 64 * 512) return;
  hb[i] = (__bf16)h0[i];
  hf[i] = h0[i];
  c[i] = c0[i];
}

// ---------------------------------------------------------------------------
// Conv1d as implicit GEMM + fused bias/maxpool4/relu -> seq (256,64,512) bf16
// Grid: (tp=256, coBlock=4), 256 threads (8 waves). Each WG: M=256 rows
// (4 conv positions x 64 batch), N=128 cout, K=2112. Wave -> one 16-col strip,
// 16 M-tiles. Pool across the 4 conv positions in registers.
// A chunks staged with async global->LDS, double-buffered; OOB taps zeroed
// with plain LDS stores.
// ---------------------------------------------------------------------------
__global__ __launch_bounds__(256) void k_conv_pool(const __bf16* __restrict__ xbf,
                                                   const __bf16* __restrict__ Bconv,
                                                   const float* __restrict__ conv_b,
                                                   __bf16* __restrict__ seq) {
  __shared__ __bf16 As[2][256 * 40];
  const int tp = blockIdx.x;
  const int cb = blockIdx.y;
  const int tid = threadIdx.x, lane = tid & 31, wave = tid >> 5;
  const int hl = lane >> 4, ln = lane & 15;
  const int co = cb * 128 + wave * 16 + ln;

  auto stage = [&](int kk, __bf16* buf) {
    const int k = kk >> 1;          // conv tap 0..32
    const int ci0 = (kk & 1) << 5;  // 0 or 32
    for (int i = tid; i < 1024; i += 256) {
      int row = i >> 2, q = i & 3;
      int dt = row >> 6, b = row & 63;
      int t_in = 8 * tp + 2 * dt + k - 16;
      __bf16* dst = buf + row * 40 + q * 8;
      if (t_in >= 0 && t_in < 2048)
        async_load_b128(dst, xbf + (size_t)t_in * 4096 + b * 64 + ci0 + q * 8);
      else
        *reinterpret_cast<uint4*>(dst) = make_uint4(0u, 0u, 0u, 0u);
    }
  };

  v8f C[16];
#pragma unroll
  for (int i = 0; i < 16; ++i) C[i] = v8f_zero();

  stage(0, As[0]);
  for (int kk = 0; kk < 66; ++kk) {   // K = 2112 in chunks of 32 (never straddles a tap)
    wait_async0();                    // this wave's chunk-kk stores landed in LDS
    __syncthreads();                  // everyone's did
    __bf16* cur = As[kk & 1];
    if (kk + 1 < 66) stage(kk + 1, As[(kk + 1) & 1]);   // overlap with WMMAs below
    v16bf bf = glb_loadB(Bconv, 2112, co, kk * 32, hl);
#pragma unroll
    for (int mt = 0; mt < 16; ++mt) {
      v16bf af = lds_loadA(cur + mt * 16 * 40, 40, ln, hl);
      C[mt] = WMMA_BF16(af, bf, C[mt]);
    }
  }

  const float bias = conv_b[co];
#pragma unroll
  for (int pm = 0; pm < 4; ++pm) {
#pragma unroll
    for (int r = 0; r < 8; ++r) {
      float v = C[pm][r];
#pragma unroll
      for (int dt = 1; dt < 4; ++dt) v = fmaxf(v, C[dt * 4 + pm][r]);   // maxpool over 4 positions
      v = fmaxf(v + bias, 0.0f);                                        // bias + relu
      int b = pm * 16 + hl * 8 + r;
      seq[((size_t)tp * 64 + b) * 512 + co] = (__bf16)v;
    }
  }
}

// ---------------------------------------------------------------------------
// wi_all = seq(16384x512) @ wIHt  -> (16384 x 2048) fp32
// Grid: (128 Mblk, 16 Nblk), 256 threads. WG tile 128x128; each wave: 1 N-strip
// x 8 M-tiles.  Async double-buffered A staging.
// ---------------------------------------------------------------------------
__global__ __launch_bounds__(256) void k_wi_gemm(const __bf16* __restrict__ seq,
                                                 const __bf16* __restrict__ wIHt,
                                                 float* __restrict__ wi_all) {
  __shared__ __bf16 As[2][128 * 40];
  const int tid = threadIdx.x, lane = tid & 31, wave = tid >> 5;
  const int hl = lane >> 4, ln = lane & 15;
  const int r0 = blockIdx.x * 128;
  const int col = blockIdx.y * 128 + wave * 16 + ln;

  auto stage = [&](int k0, __bf16* buf) {
    for (int i = tid; i < 512; i += 256) {
      int row = i >> 2, q = i & 3;
      async_load_b128(buf + row * 40 + q * 8,
                      seq + (size_t)(r0 + row) * 512 + k0 + q * 8);
    }
  };

  v8f C[8];
#pragma unroll
  for (int i = 0; i < 8; ++i) C[i] = v8f_zero();

  stage(0, As[0]);
  for (int kc = 0; kc < 16; ++kc) {
    wait_async0();
    __syncthreads();
    __bf16* cur = As[kc & 1];
    if (kc + 1 < 16) stage((kc + 1) * 32, As[(kc + 1) & 1]);
    v16bf bf = glb_loadB(wIHt, 512, col, kc * 32, hl);
#pragma unroll
    for (int mt = 0; mt < 8; ++mt) {
      v16bf af = lds_loadA(cur + mt * 16 * 40, 40, ln, hl);
      C[mt] = WMMA_BF16(af, bf, C[mt]);
    }
  }
#pragma unroll
  for (int mt = 0; mt < 8; ++mt)
#pragma unroll
    for (int r = 0; r < 8; ++r) {
      int row = r0 + mt * 16 + hl * 8 + r;
      wi_all[(size_t)row * 2048 + col] = C[mt][r];
    }
}

// ---------------------------------------------------------------------------
// One LSTM timestep. Grid: 8 WGs x 128 threads (4 waves). Each wave owns 16
// hidden columns across ALL 4 gates and ALL 64 batch rows -> batch-norm is an
// intra-wave reduction + one shfl_xor(16). h is ping-ponged across launches.
// ---------------------------------------------------------------------------
__global__ __launch_bounds__(128) void k_lstm_step(int t,
                                                   const __bf16* __restrict__ wHHt,
                                                   const float* __restrict__ wi_all,
                                                   const float* __restrict__ bias,
                                                   const float* __restrict__ big, const float* __restrict__ bib,
                                                   const float* __restrict__ bhg, const float* __restrict__ bhb,
                                                   const float* __restrict__ bcg, const float* __restrict__ bcb,
                                                   const __bf16* __restrict__ hin,
                                                   __bf16* __restrict__ hout,
                                                   float* __restrict__ hf32,
                                                   float* __restrict__ cbuf) {
  __shared__ __bf16 As[2][64 * 40];
  const int tid = threadIdx.x, lane = tid & 31, wave = tid >> 5;
  const int hl = lane >> 4, ln = lane & 15;
  const int hc = blockIdx.x * 64 + wave * 16 + ln;   // hidden column this lane owns
  const float inv64 = 1.0f / 64.0f;

  auto stage = [&](int k0, __bf16* buf) {
    for (int i = tid; i < 256; i += 128) {
      int row = i >> 2, q = i & 3;
      async_load_b128(buf + row * 40 + q * 8,
                      hin + (size_t)row * 512 + k0 + q * 8);
    }
  };

  v8f C[16];                                         // [gate*4 + mtile]
#pragma unroll
  for (int i = 0; i < 16; ++i) C[i] = v8f_zero();

  // wh = h @ weight_hh for this wave's 4 gate strips (M=64, K=512)
  stage(0, As[0]);
  for (int kc = 0; kc < 16; ++kc) {
    wait_async0();
    __syncthreads();
    __bf16* cur = As[kc & 1];
    if (kc + 1 < 16) stage((kc + 1) * 32, As[(kc + 1) & 1]);
    const int k0 = kc * 32;
    v16bf bf0 = glb_loadB(wHHt, 512, 0 * 512 + hc, k0, hl);
    v16bf bf1 = glb_loadB(wHHt, 512, 1 * 512 + hc, k0, hl);
    v16bf bf2 = glb_loadB(wHHt, 512, 2 * 512 + hc, k0, hl);
    v16bf bf3 = glb_loadB(wHHt, 512, 3 * 512 + hc, k0, hl);
#pragma unroll
    for (int mt = 0; mt < 4; ++mt) {
      v16bf af = lds_loadA(cur + mt * 16 * 40, 40, ln, hl);
      C[0 * 4 + mt] = WMMA_BF16(af, bf0, C[0 * 4 + mt]);
      C[1 * 4 + mt] = WMMA_BF16(af, bf1, C[1 * 4 + mt]);
      C[2 * 4 + mt] = WMMA_BF16(af, bf2, C[2 * 4 + mt]);
      C[3 * 4 + mt] = WMMA_BF16(af, bf3, C[3 * 4 + mt]);
    }
  }

  // a = BN(wh) + BN(wi_t) + bias, per gate (BN over batch=64, biased var, eps 1e-3)
#pragma unroll
  for (int g = 0; g < 4; ++g) {
    const int col = g * 512 + hc;
    float s = 0.f, s2 = 0.f;
#pragma unroll
    for (int mt = 0; mt < 4; ++mt)
#pragma unroll
      for (int r = 0; r < 8; ++r) { float v = C[g * 4 + mt][r]; s += v; s2 += v * v; }
    s  += __shfl_xor(s, 16, 32);
    s2 += __shfl_xor(s2, 16, 32);
    float mean = s * inv64, var = s2 * inv64 - mean * mean;
    float scl = bhg[col] * rsqrtf(var + 1e-3f);
    float sh  = bhb[col] - mean * scl;

    float w[4][8];
    float ws = 0.f, ws2 = 0.f;
#pragma unroll
    for (int mt = 0; mt < 4; ++mt)
#pragma unroll
      for (int r = 0; r < 8; ++r) {
        int m = mt * 16 + hl * 8 + r;
        float v = wi_all[((size_t)t * 64 + m) * 2048 + col];
        w[mt][r] = v; ws += v; ws2 += v * v;
      }
    ws  += __shfl_xor(ws, 16, 32);
    ws2 += __shfl_xor(ws2, 16, 32);
    float wmean = ws * inv64, wvar = ws2 * inv64 - wmean * wmean;
    float wscl = big[col] * rsqrtf(wvar + 1e-3f);
    float wsh  = bib[col] - wmean * wscl;
    float bsum = bias[col];
#pragma unroll
    for (int mt = 0; mt < 4; ++mt)
#pragma unroll
      for (int r = 0; r < 8; ++r)
        C[g * 4 + mt][r] = C[g * 4 + mt][r] * scl + sh + w[mt][r] * wscl + wsh + bsum;
  }

  // gates: c1 = sig(f)*c + sig(i)*tanh(g);  h1 = sig(o)*tanh(BN_c(c1))
  float c1[4][8];
  float s = 0.f, s2 = 0.f;
#pragma unroll
  for (int mt = 0; mt < 4; ++mt)
#pragma unroll
    for (int r = 0; r < 8; ++r) {
      int m = mt * 16 + hl * 8 + r;
      float f  = C[0 * 4 + mt][r];
      float ig = C[1 * 4 + mt][r];
      float gg = C[3 * 4 + mt][r];
      float cp = cbuf[(size_t)m * 512 + hc];
      float v = sigmoidf_(f) * cp + sigmoidf_(ig) * tanhf(gg);
      c1[mt][r] = v; s += v; s2 += v * v;
      cbuf[(size_t)m * 512 + hc] = v;
    }
  s  += __shfl_xor(s, 16, 32);
  s2 += __shfl_xor(s2, 16, 32);
  float mean = s * inv64, var = s2 * inv64 - mean * mean;
  float scl = bcg[hc] * rsqrtf(var + 1e-3f);
  float sh  = bcb[hc] - mean * scl;
#pragma unroll
  for (int mt = 0; mt < 4; ++mt)
#pragma unroll
    for (int r = 0; r < 8; ++r) {
      int m = mt * 16 + hl * 8 + r;
      float h1 = sigmoidf_(C[2 * 4 + mt][r]) * tanhf(c1[mt][r] * scl + sh);
      hout[(size_t)m * 512 + hc] = (__bf16)h1;
      hf32[(size_t)m * 512 + hc] = h1;
    }
}

// ---------------------------------------------------------------------------
// logits = h_last @ fc_w^T + fc_b; softmax over 2 classes. 64 lanes (1/batch).
// ---------------------------------------------------------------------------
__global__ __launch_bounds__(64) void k_fc_softmax(const float* __restrict__ hf,
                                                   const float* __restrict__ fcw,
                                                   const float* __restrict__ fcb,
                                                   float* __restrict__ out) {
  int b = threadIdx.x;
  float a0 = fcb[0], a1 = fcb[1];
  for (int j = 0; j < 512; ++j) {
    float hv = hf[(size_t)b * 512 + j];
    a0 += hv * fcw[j];
    a1 += hv * fcw[512 + j];
  }
  float m = fmaxf(a0, a1);
  float e0 = __expf(a0 - m), e1 = __expf(a1 - m);
  float inv = 1.0f / (e0 + e1);
  out[b * 2 + 0] = e0 * inv;
  out[b * 2 + 1] = e1 * inv;
}

// ---------------------------------------------------------------------------
extern "C" void kernel_launch(void* const* d_in, const int* in_sizes, int n_in,
                              void* d_out, int out_size, void* d_ws, size_t ws_size,
                              hipStream_t stream) {
  (void)in_sizes; (void)n_in; (void)out_size; (void)ws_size;
  const float* x      = (const float*)d_in[0];
  const float* conv_w = (const float*)d_in[1];
  const float* conv_b = (const float*)d_in[2];
  const float* w_ih   = (const float*)d_in[3];
  const float* w_hh   = (const float*)d_in[4];
  const float* bias   = (const float*)d_in[5];
  const float* big    = (const float*)d_in[6];
  const float* bib    = (const float*)d_in[7];
  const float* bhg    = (const float*)d_in[8];
  const float* bhb    = (const float*)d_in[9];
  const float* bcg    = (const float*)d_in[10];
  const float* bcb    = (const float*)d_in[11];
  const float* fcw    = (const float*)d_in[12];
  const float* fcb    = (const float*)d_in[13];
  const float* h0     = (const float*)d_in[14];
  const float* c0     = (const float*)d_in[15];
  float* out = (float*)d_out;

  char* wsb = (char*)d_ws;
  size_t off = 0;
  auto carve = [&](size_t bytes) -> char* {
    char* p = wsb + off;
    off += (bytes + 255) & ~(size_t)255;
    return p;
  };
  __bf16* xbf   = (__bf16*)carve((size_t)2048 * 64 * 64 * 2);   // 16.8 MB
  __bf16* Bconv = (__bf16*)carve((size_t)512 * 2112 * 2);       //  2.2 MB
  __bf16* wIHt  = (__bf16*)carve((size_t)2048 * 512 * 2);       //  2.1 MB
  __bf16* wHHt  = (__bf16*)carve((size_t)2048 * 512 * 2);       //  2.1 MB
  __bf16* seq   = (__bf16*)carve((size_t)256 * 64 * 512 * 2);   // 16.8 MB
  float*  wiAll = (float*) carve((size_t)16384 * 2048 * 4);     // 134 MB
  __bf16* hb0   = (__bf16*)carve((size_t)64 * 512 * 2);
  __bf16* hb1   = (__bf16*)carve((size_t)64 * 512 * 2);
  float*  hf    = (float*) carve((size_t)64 * 512 * 4);
  float*  cbuf  = (float*) carve((size_t)64 * 512 * 4);

  // prep
  k_cvt_x<<<(2048 * 64 * 64 + 255) / 256, 256, 0, stream>>>(x, xbf, 2048 * 64 * 64);
  k_pack_convw<<<(512 * 2112 + 255) / 256, 256, 0, stream>>>(conv_w, Bconv);
  k_pack_wt<<<(2048 * 512 + 255) / 256, 256, 0, stream>>>(w_ih, wIHt);
  k_pack_wt<<<(2048 * 512 + 255) / 256, 256, 0, stream>>>(w_hh, wHHt);
  k_init_hc<<<(64 * 512 + 255) / 256, 256, 0, stream>>>(h0, c0, hb0, hf, cbuf);

  // conv + pool + relu -> seq
  k_conv_pool<<<dim3(256, 4), 256, 0, stream>>>(xbf, Bconv, conv_b, seq);

  // wi_all = seq @ W_ih
  k_wi_gemm<<<dim3(128, 16), 256, 0, stream>>>(seq, wIHt, wiAll);

  // recurrent scan (sequential launches; h ping-pong)
  for (int t = 0; t < 256; ++t) {
    const __bf16* hin = (t & 1) ? hb1 : hb0;
    __bf16* hout      = (t & 1) ? hb0 : hb1;
    k_lstm_step<<<8, 128, 0, stream>>>(t, wHHt, wiAll, bias, big, bib, bhg, bhb,
                                       bcg, bcb, hin, hout, hf, cbuf);
  }

  // classifier
  k_fc_softmax<<<1, 64, 0, stream>>>(hf, fcw, fcb, out);
}
